// Sequence_17712445129091
// MI455X (gfx1250) — matrix operand order repaired and checked by reading, not compiled, not run on hardware
//
#include <hip/hip_runtime.h>

// LSTM encoder-decoder, B=4096, T=64, H=128 on gfx1250 (MI455X).
//  - prep:      fold biases, convert Whh_e / Whh_d to bf16 in workspace.
//  - enc:       64 recurrent steps; per block: 16 batch rows. The per-wave
//               B fragments of Whh_e (16 frags = 128 VGPRs) are LOOP-INVARIANT
//               and held in registers across all steps; only h recirculates
//               through LDS (4KB). Gates via v_wmma_f32_16x16x32_bf16 with
//               f32 accumulate. Wave w owns h-columns [16w,16w+16), so the
//               i/f/g/o nonlinearity combine is thread-local; c stays in VGPRs.
//               LDS/block ~8.3KB -> high occupancy for latency hiding.
//  - dec_gemm:  one step of the same GEMM: hpart = h_e @ Whh_d^T + b_d.
//  - dec_out:   fused elementwise decoder + GEMV + 63x repeated output writes
//               (memory-bound: ~66MB of output ~ 3us at 23.3 TB/s).

#define HN   128
#define TN   64
#define BN   4096
#define G4   512          // 4*H
#define OUTW 4033         // 1 + T*(T-1)

typedef __attribute__((ext_vector_type(16))) __bf16 v16bf;
typedef __attribute__((ext_vector_type(8)))  __bf16 v8bf;
typedef __attribute__((ext_vector_type(8)))  float  v8f;

union V16U { v16bf v; v8bf h[2]; };

__device__ __forceinline__ float fsig(float x) {
    return 1.0f / (1.0f + __expf(-x));
}
__device__ __forceinline__ float ftanh(float x) {
    x = fminf(fmaxf(x, -15.0f), 15.0f);
    float e = __expf(2.0f * x);
    return (e - 1.0f) / (e + 1.0f);
}

// ---------------------------------------------------------------- prep ----
__global__ void __launch_bounds__(256)
prep_kernel(const float* __restrict__ WhhE, const float* __restrict__ WhhD,
            const float* __restrict__ bihE, const float* __restrict__ bhhE,
            const float* __restrict__ bihD, const float* __restrict__ bhhD,
            __bf16* __restrict__ whhE_bf, __bf16* __restrict__ whhD_bf,
            float* __restrict__ be, float* __restrict__ bd)
{
    int i = blockIdx.x * blockDim.x + threadIdx.x;
    if (i < G4 * HN) {
        whhE_bf[i] = (__bf16)WhhE[i];
        whhD_bf[i] = (__bf16)WhhD[i];
    }
    if (i < G4) {
        be[i] = bihE[i] + bhhE[i];
        bd[i] = bihD[i] + bhhD[i];
    }
}

// Load this lane's B fragment (32x16 bf16 tile of Whh^T) straight from
// global memory. Lane lc holds column n = col0+lc; lo half-wave takes
// K {kb..kb+7, kb+16..kb+23}, hi half-wave {kb+8..kb+15, kb+24..kb+31} (ISA B layout).
__device__ __forceinline__ V16U load_bfrag(const __bf16* __restrict__ W,
                                           int col0, int lc, int hi, int kb)
{
    V16U b;
    const __bf16* q = W + (size_t)(col0 + lc) * HN + kb + hi * 8;
    b.h[0] = *(const v8bf*)(q);
    b.h[1] = *(const v8bf*)(q + 16);
    return b;
}

// ------------------------------------------------------------- encoder ----
// grid: B/16 blocks of 256 threads (8 waves).
__global__ void __launch_bounds__(256)
enc_kernel(const float* __restrict__ input,     // [B,T]
           const float* __restrict__ WihE,      // [512]
           const __bf16* __restrict__ whhE_bf,  // ws bf16 [512][128]
           const float* __restrict__ be,        // ws [512]
           const float* __restrict__ WfcE,      // [128]
           const float* __restrict__ bfcE,      // [1]
           __bf16* __restrict__ hE,             // ws [B][128] bf16
           float*  __restrict__ cE,             // ws [B][128] f32
           float*  __restrict__ out)            // [B][4033]
{
    __shared__ __bf16 hA[16 * HN];   // current h, bf16, recirculated
    __shared__ float  xs[TN * 16];   // block's input slice, time-major
    __shared__ float  enc[16];       // fc_enc accumulator

    const int tid  = threadIdx.x;
    const int lane = tid & 31;
    const int wave = tid >> 5;
    const int hi   = lane >> 4;      // half-wave select
    const int lc   = lane & 15;
    const int m0   = blockIdx.x * 16;

    // input slice -> LDS, xs[t*16 + m]
    #pragma unroll
    for (int k = 0; k < 4; ++k) {
        int i = tid + 256 * k;
        int t = i >> 4, r = i & 15;
        xs[i] = input[(size_t)(m0 + r) * TN + t];
    }
    // h0 = 0
    #pragma unroll
    for (int k = 0; k < 8; ++k) hA[tid + 256 * k] = (__bf16)0.0f;
    if (tid < 16) enc[tid] = bfcE[0];

    // Loop-invariant per-wave state: B fragments of Whh_e + gate constants.
    // Wave w owns h columns [16w, 16w+16); its 4 gate tiles sit at
    // columns g*128 + 16w. 16 fragments = 128 VGPRs, register-resident.
    V16U bf[4][4];
    float bias[4], wih[4];
    #pragma unroll
    for (int g = 0; g < 4; ++g) {
        int col0 = g * HN + wave * 16;
        #pragma unroll
        for (int kc = 0; kc < 4; ++kc)
            bf[g][kc] = load_bfrag(whhE_bf, col0, lc, hi, kc * 32);
        bias[g] = be[col0 + lc];
        wih[g]  = WihE[col0 + lc];
    }
    float cc[8];
    #pragma unroll
    for (int r = 0; r < 8; ++r) cc[r] = 0.0f;

    __syncthreads();

    const int nh = wave * 16 + lc;   // owned h column
    for (int t = 0; t < TN; ++t) {
        // A fragments (16x32 bf16, ISA layout): lanes 0-15 take K {0..7,16..23},
        // lanes 16-31 take K {8..15,24..31} of each 32-chunk; row M = lc.
        V16U a[4];
        const __bf16* ap = hA + lc * HN + hi * 8;
        #pragma unroll
        for (int kc = 0; kc < 4; ++kc) {
            const __bf16* p = ap + kc * 32;
            a[kc].h[0] = *(const v8bf*)(p);
            a[kc].h[1] = *(const v8bf*)(p + 16);
        }

        float gv[4][8];
        #pragma unroll
        for (int g = 0; g < 4; ++g) {
            v8f acc = {};
            #pragma unroll
            for (int kc = 0; kc < 4; ++kc)
                acc = __builtin_amdgcn_wmma_f32_16x16x32_bf16(
                        false, a[kc].v, false, bf[g][kc].v,
                        (short)0, acc, false, false);
            #pragma unroll
            for (int r = 0; r < 8; ++r) {
                int m = r + 8 * hi;          // C/D layout: hi lanes hold M=8..15
                gv[g][r] = acc[r] + xs[t * 16 + m] * wih[g] + bias[g];
            }
        }

        __syncthreads();                     // all A-frag reads done
        #pragma unroll
        for (int r = 0; r < 8; ++r) {
            float cn = fsig(gv[1][r]) * cc[r] + fsig(gv[0][r]) * ftanh(gv[2][r]);
            float hn = fsig(gv[3][r]) * ftanh(cn);
            cc[r] = cn;
            int m = r + 8 * hi;
            hA[m * HN + nh] = (__bf16)hn;
        }
        __syncthreads();                     // h update visible
    }

    // enc_out = c_e @ Wfc_e^T + bfc_e ; persist h_e (bf16), c_e (f32)
    {
        float wf = WfcE[nh];
        #pragma unroll
        for (int r = 0; r < 8; ++r) {
            int m = r + 8 * hi;
            atomicAdd(&enc[m], cc[r] * wf);
            cE[(size_t)(m0 + m) * HN + nh] = cc[r];
        }
    }
    __syncthreads();
    {   // 2048 bf16 = 256 x 16B
        const uint4* ls = (const uint4*)hA;
        uint4* gd = (uint4*)(hE + (size_t)m0 * HN);
        gd[tid] = ls[tid];
    }
    if (tid < 16) out[(size_t)(m0 + tid) * OUTW] = enc[tid];
}

// ------------------------------------------------- decoder GEMM (hpart) ----
// hpart = h_e @ Whh_d^T + b_d   -> [B,512] f32. One step of the same tiling;
// B fragments straight from global (L2-hot), only h_e staged in LDS.
__global__ void __launch_bounds__(256)
dec_gemm_kernel(const __bf16* __restrict__ hE,       // ws [B][128] bf16
                const __bf16* __restrict__ whhD_bf,  // ws [512][128] bf16
                const float*  __restrict__ bd,       // ws [512]
                float* __restrict__ hpart)           // ws [B][512]
{
    __shared__ __bf16 hA[16 * HN];

    const int tid  = threadIdx.x;
    const int lane = tid & 31;
    const int wave = tid >> 5;
    const int hi   = lane >> 4;
    const int lc   = lane & 15;
    const int m0   = blockIdx.x * 16;

    {
        const uint4* gs = (const uint4*)(hE + (size_t)m0 * HN);
        uint4* ls = (uint4*)hA;
        ls[tid] = gs[tid];
    }
    __syncthreads();

    V16U a[4];
    const __bf16* ap = hA + lc * HN + hi * 8;
    #pragma unroll
    for (int kc = 0; kc < 4; ++kc) {
        const __bf16* p = ap + kc * 32;
        a[kc].h[0] = *(const v8bf*)(p);
        a[kc].h[1] = *(const v8bf*)(p + 16);
    }
    #pragma unroll
    for (int g = 0; g < 4; ++g) {
        int col0 = g * HN + wave * 16;
        v8f acc = {};
        #pragma unroll
        for (int kc = 0; kc < 4; ++kc) {
            V16U b = load_bfrag(whhD_bf, col0, lc, hi, kc * 32);
            acc = __builtin_amdgcn_wmma_f32_16x16x32_bf16(
                    false, a[kc].v, false, b.v, (short)0, acc, false, false);
        }
        float bb = bd[col0 + lc];
        #pragma unroll
        for (int r = 0; r < 8; ++r) {
            int m = r + 8 * hi;
            hpart[(size_t)(m0 + m) * G4 + col0 + lc] = acc[r] + bb;
        }
    }
}

// ----------------------------------------- fused decoder + output writes ----
// One block per batch row. dec_out(m,t) repeated 63x into out[m, 1+63t .. ].
__global__ void __launch_bounds__(256)
dec_out_kernel(const float* __restrict__ input_rev,  // [B,T]
               const float* __restrict__ hpart,      // ws [B,512]
               const float* __restrict__ cE,         // ws [B,128]
               const float* __restrict__ WihD,       // [512]
               const float* __restrict__ WfcD,       // [128]
               const float* __restrict__ bfcD,       // [1]
               float* __restrict__ out)              // [B,4033]
{
    __shared__ float shp[G4], swih[G4], sce[HN], swfc[HN];
    const int m    = blockIdx.x;
    const int tid  = threadIdx.x;
    const int lane = tid & 31;
    const int wave = tid >> 5;

    for (int i = tid; i < G4; i += 256) {
        shp[i]  = hpart[(size_t)m * G4 + i];
        swih[i] = WihD[i];
    }
    for (int i = tid; i < HN; i += 256) {
        sce[i]  = cE[(size_t)m * HN + i];
        swfc[i] = WfcD[i];
    }
    const float bfc = bfcD[0];
    __syncthreads();

    for (int t = wave; t < TN; t += 8) {
        float x = input_rev[(size_t)m * TN + t];
        float s = 0.0f;
        #pragma unroll
        for (int j = 0; j < 4; ++j) {
            int h = j * 32 + lane;
            float gi = x * swih[h]          + shp[h];
            float gf = x * swih[HN + h]     + shp[HN + h];
            float gg = x * swih[2 * HN + h] + shp[2 * HN + h];
            float c  = fsig(gf) * sce[h] + fsig(gi) * ftanh(gg);
            s += c * swfc[h];
        }
        #pragma unroll
        for (int off = 16; off > 0; off >>= 1) s += __shfl_xor(s, off, 32);
        float val = s + bfc;
        size_t base = (size_t)m * OUTW + 1 + (size_t)t * 63;
        for (int j = lane; j < 63; j += 32) out[base + j] = val;
    }
}

// ------------------------------------------------------------- launcher ----
extern "C" void kernel_launch(void* const* d_in, const int* in_sizes, int n_in,
                              void* d_out, int out_size, void* d_ws, size_t ws_size,
                              hipStream_t stream)
{
    const float* input     = (const float*)d_in[0];
    const float* input_rev = (const float*)d_in[1];
    const float* WihE      = (const float*)d_in[2];
    const float* WhhE      = (const float*)d_in[3];
    const float* bihE      = (const float*)d_in[4];
    const float* bhhE      = (const float*)d_in[5];
    const float* WfcE      = (const float*)d_in[6];
    const float* bfcE      = (const float*)d_in[7];
    const float* WihD      = (const float*)d_in[8];
    const float* WhhD      = (const float*)d_in[9];
    const float* bihD      = (const float*)d_in[10];
    const float* bhhD      = (const float*)d_in[11];
    const float* WfcD      = (const float*)d_in[12];
    const float* bfcD      = (const float*)d_in[13];
    float* out = (float*)d_out;

    // workspace layout (16B aligned)
    char* ws = (char*)d_ws;
    __bf16* whhE_bf = (__bf16*)(ws + 0);         // 131072
    __bf16* whhD_bf = (__bf16*)(ws + 131072);    // 131072
    float*  be      = (float*) (ws + 262144);    // 2048
    float*  bd      = (float*) (ws + 264192);    // 2048
    __bf16* hE      = (__bf16*)(ws + 266240);    // 1048576
    float*  cE      = (float*) (ws + 1314816);   // 2097152
    float*  hpart   = (float*) (ws + 3411968);   // 8388608

    prep_kernel<<<(G4 * HN + 255) / 256, 256, 0, stream>>>(
        WhhE, WhhD, bihE, bhhE, bihD, bhhD, whhE_bf, whhD_bf, be, bd);

    enc_kernel<<<BN / 16, 256, 0, stream>>>(
        input, WihE, whhE_bf, be, WfcE, bfcE, hE, cE, out);

    dec_gemm_kernel<<<BN / 16, 256, 0, stream>>>(
        hE, whhD_bf, bd, hpart);

    dec_out_kernel<<<BN, 256, 0, stream>>>(
        input_rev, hpart, cE, WihD, WfcD, bfcD, out);
}
